// PointPillarEncoder_57397942944144
// MI455X (gfx1250) — matrix operand order repaired and checked by reading, not compiled, not run on hardware
//
#include <hip/hip_runtime.h>
#include <hip/hip_bf16.h>

// ---------------------------------------------------------------------------
// PointPillars-style pipeline for MI455X (gfx1250), bf16 WMMA implicit GEMM.
// ---------------------------------------------------------------------------

#define NVOX   150000
#define XDIM   512
#define YDIM   512
#define NSP    (XDIM*YDIM)     // 262144 spatial positions
#define EPSBN  1e-5f

typedef __attribute__((ext_vector_type(16))) __bf16 v16bf;
typedef __attribute__((ext_vector_type(8)))  float  v8f;
typedef int v4i __attribute__((vector_size(16)));

struct alignas(16) U128 { unsigned int a, b, c, d; };
union FragU { v16bf v; U128 q[2]; };

#if __has_builtin(__builtin_amdgcn_global_load_async_to_lds_b128)
#define HAS_ASYNC_LDS 1
#else
#define HAS_ASYNC_LDS 0
#endif

// 16B global -> LDS copy: async (ASYNCcnt-tracked, no VGPR round trip) if available.
__device__ inline void copy16_g2l(__bf16* l, const __bf16* g) {
#if HAS_ASYNC_LDS
    __builtin_amdgcn_global_load_async_to_lds_b128((v4i*)g, (v4i*)l, 0, 0);
#else
    *(U128*)l = *(const U128*)g;
#endif
}
__device__ inline void wait_async_copies() {
#if HAS_ASYNC_LDS
  #if __has_builtin(__builtin_amdgcn_s_wait_asynccnt)
    __builtin_amdgcn_s_wait_asynccnt(0);
  #else
    asm volatile("s_wait_asynccnt 0x0" ::: "memory");
  #endif
#endif
}

// A fragment (16x32 bf16): rowPtr -> K=0 of this lane's M row (32 contiguous K)
__device__ inline v16bf load_frag_a(const __bf16* rowPtr, int lane) {
    const int koff = (lane < 16) ? 0 : 8;
    FragU f;
    f.q[0] = *(const U128*)(rowPtr + koff);
    f.q[1] = *(const U128*)(rowPtr + koff + 16);
    return f.v;
}
// B fragment (32x16 bf16): colPtr -> K=0 of this lane's N column (32 contiguous K)
__device__ inline v16bf load_frag_b(const __bf16* colPtr, int lane) {
    const int koff = (lane < 16) ? 0 : 8;
    FragU f;
    f.q[0] = *(const U128*)(colPtr + koff);
    f.q[1] = *(const U128*)(colPtr + koff + 16);
    return f.v;
}

// ---------------------------------------------------------------------------
__global__ void zero_kernel(U128* __restrict__ p, int n) {
    int i = blockIdx.x * blockDim.x + threadIdx.x;
    if (i < n) { U128 z = {0u,0u,0u,0u}; p[i] = z; }
}

// convA_w/convB_w (OIHW f32) -> wt[(o*9+tap)*256 + cin] bf16
__global__ void convert_w3_kernel(const float* __restrict__ w, __bf16* __restrict__ wt) {
    int i = blockIdx.x * 256 + threadIdx.x;      // over 256*256*9
    if (i >= 256 * 256 * 9) return;
    int o   = i / 2304;
    int rem = i % 2304;
    int cin = rem / 9;
    int tap = rem % 9;
    wt[(o * 9 + tap) * 256 + cin] = (__bf16)w[i];
}

// ---------------------------------------------------------------------------
// PFN: layer1 (4->64) on VALU into LDS bf16, layer2 (64->64 over 128 rows =
// 4 pillars x 32 points) on WMMA, ReLU + mean-pool via ds_add_f32, scatter.
__global__ __launch_bounds__(128)
void pfn_wmma_kernel(const float* __restrict__ pillars, const int* __restrict__ num_points,
                     const int* __restrict__ indices,
                     const float* __restrict__ w1, const float* __restrict__ b1,
                     const float* __restrict__ w2, const float* __restrict__ b2,
                     __bf16* __restrict__ bev) {
    __shared__ float sW1[256];
    __shared__ float sB1[64];
    __shared__ float sB2[64];
    __shared__ __align__(16) __bf16 sH[128 * 64];   // layer1 activations (rows x h)
    __shared__ __align__(16) __bf16 sW2[64 * 64];   // w2 as [g][h] bf16
    __shared__ float sPool[4 * 64];                 // per (pillar, channel) sums
    const int tid = threadIdx.x, lane = tid & 31, wv = tid >> 5;
    const int vbase = blockIdx.x * 4;               // 4 pillars per block

    for (int i = tid; i < 256; i += 128) { sW1[i] = w1[i]; sPool[i] = 0.0f; }
    if (tid < 64) { sB1[tid] = b1[tid]; sB2[tid] = b2[tid]; }
    for (int i = tid; i < 4096; i += 128) sW2[i] = (__bf16)w2[i];
    __syncthreads();

    // layer1: one row (pillar, point) per thread
    {
        const int v = vbase + (tid >> 5);
        const int p = tid & 31;
        const float maskf = (num_points[v] > 0) ? 1.0f : 0.0f;
        float4 xc = *(const float4*)(pillars + (long long)v * 128 + p * 4);
        const float x0 = xc.x * maskf, x1 = xc.y * maskf;
        const float x2 = xc.z * maskf, x3 = xc.w * maskf;
        __bf16* hrow = sH + tid * 64;
        for (int h = 0; h < 64; ++h) {
            float s = sB1[h] + sW1[h*4+0]*x0 + sW1[h*4+1]*x1
                             + sW1[h*4+2]*x2 + sW1[h*4+3]*x3;
            hrow[h] = (__bf16)fmaxf(s, 0.0f);
        }
    }
    __syncthreads();

    // layer2 GEMM: M=128 rows, N=64, K=64. Wave wv owns N subtile wv*16..+15.
    v8f acc[8];
    const v8f vzero = {0.f,0.f,0.f,0.f,0.f,0.f,0.f,0.f};
    #pragma unroll
    for (int i = 0; i < 8; ++i) acc[i] = vzero;

    for (int ks = 0; ks < 2; ++ks) {
        v16bf bf = load_frag_b(sW2 + (wv * 16 + (lane & 15)) * 64 + ks * 32, lane);
        #pragma unroll
        for (int mt = 0; mt < 8; ++mt) {
            v16bf af = load_frag_a(sH + (mt * 16 + (lane & 15)) * 64 + ks * 32, lane);
            acc[mt] = __builtin_amdgcn_wmma_f32_16x16x32_bf16(
                false, af, false, bf, (short)0, acc[mt], false, false);
        }
    }
    // epilogue: +b2, ReLU, pool (tile mt belongs to pillar mt>>1)
    const int n = wv * 16 + (lane & 15);
    const float b2v = sB2[n];
    #pragma unroll
    for (int mt = 0; mt < 8; ++mt) {
        float s = 0.0f;
        #pragma unroll
        for (int e = 0; e < 8; ++e) s += fmaxf(acc[mt][e] + b2v, 0.0f);
        atomicAdd(&sPool[(mt >> 1) * 64 + n], s);
    }
    __syncthreads();

    // scatter: 4 pillars x 64 ch, mean over 32 points
    for (int i = tid; i < 256; i += 128) {
        const int p = i >> 6, c = i & 63;
        const int v = vbase + p;
        const int xs = indices[v * 2 + 0];
        const int ys = indices[v * 2 + 1];
        if ((unsigned)xs < (unsigned)XDIM && (unsigned)ys < (unsigned)YDIM)
            bev[((((xs << 9) + ys)) << 6) + c] = (__bf16)(sPool[i] * (1.0f / 32.0f));
    }
}

// ---------------------------------------------------------------------------
// 1x1 conv (64 -> 256) as WMMA GEMM: M=128 spatial, N=64 cout per block.
__global__ __launch_bounds__(128)
void conv1x1_wmma(const __bf16* __restrict__ in,   // [512][512][64]
                  const float*  __restrict__ w,    // [256][64] f32
                  const float*  __restrict__ bias, // [256]
                  __bf16* __restrict__ out) {      // [512][512][256]
    __shared__ __align__(16) __bf16 sA[128 * 64];
    __shared__ __align__(16) __bf16 sB[64 * 64];
    const int bid = blockIdx.x;
    const int n64 = (bid & 3) * 64;
    const int y0  = ((bid >> 2) & 3) * 128;
    const int x   = bid >> 4;
    const int tid = threadIdx.x, lane = tid & 31, wv = tid >> 5;

    for (int i = tid; i < 1024; i += 128) {         // 128 rows x 8 chunks of 16B
        int m = i >> 3, c8 = (i & 7) << 3;
        copy16_g2l(sA + m * 64 + c8, in + ((((x << 9) + y0 + m) << 6) + c8));
    }
    for (int i = tid; i < 64 * 64; i += 128) {
        int nn = i >> 6, k = i & 63;
        sB[nn * 64 + k] = (__bf16)w[(n64 + nn) * 64 + k];
    }
    wait_async_copies();
    __syncthreads();

    v8f acc[8];
    const v8f vzero = {0.f,0.f,0.f,0.f,0.f,0.f,0.f,0.f};
    #pragma unroll
    for (int i = 0; i < 8; ++i) acc[i] = vzero;

    for (int ks = 0; ks < 2; ++ks) {
        v16bf bf = load_frag_b(sB + (wv * 16 + (lane & 15)) * 64 + ks * 32, lane);
        #pragma unroll
        for (int mt = 0; mt < 8; ++mt) {
            v16bf af = load_frag_a(sA + (mt * 16 + (lane & 15)) * 64 + ks * 32, lane);
            acc[mt] = __builtin_amdgcn_wmma_f32_16x16x32_bf16(
                false, af, false, bf, (short)0, acc[mt], false, false);
        }
    }
    const int n = n64 + wv * 16 + (lane & 15);
    const float bv = bias[n];
    #pragma unroll
    for (int mt = 0; mt < 8; ++mt) {
        const int ybase = y0 + mt * 16 + ((lane < 16) ? 0 : 8);
        #pragma unroll
        for (int e = 0; e < 8; ++e)
            out[((((x << 9) + ybase + e) << 8)) + n] = (__bf16)(acc[mt][e] + bv);
    }
}

// ---------------------------------------------------------------------------
// 3x3 conv (256 -> 256) implicit GEMM. Block: M=128 spatial (one x-row), N=64 cout.
// K loop: 8 cin-blocks of 32 x 9 taps. Slabs staged via async global->LDS copies.
__global__ __launch_bounds__(128)
void conv3x3_wmma(const __bf16* __restrict__ in,   // [512][512][256] NHWC bf16
                  const __bf16* __restrict__ wt,   // [256][9][256] (o,tap,cin) bf16
                  const float*  __restrict__ bias, // [256]
                  __bf16* __restrict__ out,        // [512][512][256]
                  float* __restrict__ ssum, float* __restrict__ ssq) {
    __shared__ __align__(16) __bf16 sA[3 * 130 * 32];  // 24.96 KB
    __shared__ __align__(16) __bf16 sB[64 * 9 * 32];   // 36.86 KB
    const int bid = blockIdx.x;
    const int n64 = (bid & 3) * 64;
    const int y0  = ((bid >> 2) & 3) * 128;
    const int x   = bid >> 4;
    const int tid = threadIdx.x, lane = tid & 31, wv = tid >> 5;

    v8f acc[8];
    const v8f vzero = {0.f,0.f,0.f,0.f,0.f,0.f,0.f,0.f};
    #pragma unroll
    for (int i = 0; i < 8; ++i) acc[i] = vzero;

    for (int cb = 0; cb < 8; ++cb) {
        __syncthreads();
        // A slab: 3 input rows x 130 y x 32 cin (zero padded at borders)
        for (int i = tid; i < 3 * 130; i += 128) {
            const int r = i / 130, yc = i % 130;
            const int xx = x + r - 1;
            const int yy = y0 + yc - 1;
            __bf16* dst = sA + i * 32;
            if ((unsigned)xx < (unsigned)XDIM && (unsigned)yy < (unsigned)YDIM) {
                const __bf16* src = in + ((((xx << 9) + yy) << 8) + (cb << 5));
                copy16_g2l(dst, src);
                copy16_g2l(dst + 8, src + 8);
            } else {
                U128 z = {0u,0u,0u,0u};
                ((U128*)dst)[0] = z;
                ((U128*)dst)[1] = z;
            }
        }
        // B slab: 64 cout x 9 taps x 32 cin (bf16, pre-transposed weights)
        for (int i = tid; i < 64 * 9 * 2; i += 128) {
            const int ot = i >> 1, half = i & 1;
            const int o = ot / 9, tap = ot % 9;
            copy16_g2l(sB + ot * 32 + half * 8,
                       wt + (((n64 + o) * 9 + tap) * 256 + (cb << 5) + half * 8));
        }
        wait_async_copies();
        __syncthreads();

        for (int tap = 0; tap < 9; ++tap) {
            const int r = tap / 3, dy = tap % 3;
            v16bf bf = load_frag_b(sB + ((wv * 16 + (lane & 15)) * 9 + tap) * 32, lane);
            #pragma unroll
            for (int mt = 0; mt < 8; ++mt) {
                const __bf16* rowPtr = sA + ((r * 130) + (mt * 16 + (lane & 15)) + dy) * 32;
                v16bf af = load_frag_a(rowPtr, lane);
                acc[mt] = __builtin_amdgcn_wmma_f32_16x16x32_bf16(
                    false, af, false, bf, (short)0, acc[mt], false, false);
            }
        }
    }
    // Epilogue: bias, bf16 store, BN partial sums
    const int n = n64 + wv * 16 + (lane & 15);
    const float bv = bias[n];
    float lsum = 0.f, lsq = 0.f;
    #pragma unroll
    for (int mt = 0; mt < 8; ++mt) {
        const int ybase = y0 + mt * 16 + ((lane < 16) ? 0 : 8);
        #pragma unroll
        for (int e = 0; e < 8; ++e) {
            const float val = acc[mt][e] + bv;
            lsum += val; lsq += val * val;
            out[((((x << 9) + ybase + e) << 8)) + n] = (__bf16)val;
        }
    }
    atomicAdd(&ssum[n], lsum);
    atomicAdd(&ssq[n],  lsq);
}

// ---------------------------------------------------------------------------
// BN + ReLU, bf16 -> bf16 (in-place ok). NHWC.
__global__ __launch_bounds__(256)
void bn_relu_bf16_kernel(const __bf16* __restrict__ in, __bf16* __restrict__ out,
                         const float* __restrict__ ssum, const float* __restrict__ ssq,
                         const float* __restrict__ g, const float* __restrict__ beta) {
    const int idx = blockIdx.x * 256 + threadIdx.x;   // NSP * 32 units of 8 channels
    if (idx >= NSP * 32) return;
    const int c0   = (idx & 31) << 3;
    const int base = ((idx >> 5) << 8) + c0;
    const float invN = 1.0f / (float)NSP;
    U128 d = *(const U128*)(in + base);
    const __bf16* dp = (const __bf16*)&d;
    U128 o;
    __bf16* op = (__bf16*)&o;
    #pragma unroll
    for (int i = 0; i < 8; ++i) {
        const int c = c0 + i;
        const float mean  = ssum[c] * invN;
        const float var   = ssq[c] * invN - mean * mean;
        const float scale = g[c] * rsqrtf(var + EPSBN);
        const float val   = ((float)dp[i] - mean) * scale + beta[c];
        op[i] = (__bf16)fmaxf(val, 0.0f);
    }
    *(U128*)(out + base) = o;
}

// BN + ReLU final: NHWC bf16 -> NCHW f32 (d_out), transposed through LDS.
__global__ __launch_bounds__(256)
void bn_relu_final_kernel(const __bf16* __restrict__ in, float* __restrict__ out,
                          const float* __restrict__ ssum, const float* __restrict__ ssq,
                          const float* __restrict__ g, const float* __restrict__ beta) {
    __shared__ float tile[32][257];
    const int s0  = blockIdx.x * 32;
    const int tid = threadIdx.x;
    const float invN = 1.0f / (float)NSP;
    for (int i = tid; i < 32 * 32; i += 256) {       // 32 spatial x 32 chunks of 8 ch
        const int s  = i >> 5;
        const int c0 = (i & 31) << 3;
        U128 d = *(const U128*)(in + ((s0 + s) << 8) + c0);
        const __bf16* dp = (const __bf16*)&d;
        #pragma unroll
        for (int k = 0; k < 8; ++k) {
            const int c = c0 + k;
            const float mean  = ssum[c] * invN;
            const float var   = ssq[c] * invN - mean * mean;
            const float scale = g[c] * rsqrtf(var + EPSBN);
            const float val   = ((float)dp[k] - mean) * scale + beta[c];
            tile[s][c] = fmaxf(val, 0.0f);
        }
    }
    __syncthreads();
    for (int i = tid; i < 32 * 256; i += 256) {      // coalesced NCHW writes
        const int c = i >> 5;
        const int s = i & 31;
        out[c * NSP + s0 + s] = tile[s][c];
    }
}

// ---------------------------------------------------------------------------
// Workspace layout (bytes)
#define OFF_BEV   0LL
#define OFF_T1    33554432LL                  // 64ch bev = 33.5MB
#define OFF_TA    (OFF_T1 + 134217728LL)      // t1 = 134MB
#define OFF_STATS (OFF_TA + 134217728LL)      // tA = 134MB
#define OFF_WTA   (OFF_STATS + 4096LL)
#define OFF_WTB   (OFF_WTA + 1179648LL)

extern "C" void kernel_launch(void* const* d_in, const int* in_sizes, int n_in,
                              void* d_out, int out_size, void* d_ws, size_t ws_size,
                              hipStream_t stream) {
    (void)in_sizes; (void)n_in; (void)out_size; (void)ws_size;
    const float* pillars    = (const float*)d_in[0];
    const int*   num_points = (const int*)  d_in[1];
    const int*   indices    = (const int*)  d_in[2];
    const float* w1      = (const float*)d_in[3];
    const float* b1      = (const float*)d_in[4];
    const float* w2      = (const float*)d_in[5];
    const float* b2      = (const float*)d_in[6];
    const float* conv1_w = (const float*)d_in[7];
    const float* conv1_b = (const float*)d_in[8];
    const float* convA_w = (const float*)d_in[9];
    const float* convA_b = (const float*)d_in[10];
    const float* bnA_g   = (const float*)d_in[11];
    const float* bnA_b   = (const float*)d_in[12];
    const float* convB_w = (const float*)d_in[13];
    const float* convB_b = (const float*)d_in[14];
    const float* bnB_g   = (const float*)d_in[15];
    const float* bnB_b   = (const float*)d_in[16];

    char* ws = (char*)d_ws;
    __bf16* bev   = (__bf16*)(ws + OFF_BEV);
    __bf16* t1    = (__bf16*)(ws + OFF_T1);
    __bf16* tA    = (__bf16*)(ws + OFF_TA);
    float*  stats = (float*) (ws + OFF_STATS);
    __bf16* wtA   = (__bf16*)(ws + OFF_WTA);
    __bf16* wtB   = (__bf16*)(ws + OFF_WTB);
    float* sumA = stats + 0;   float* sqA = stats + 256;
    float* sumB = stats + 512; float* sqB = stats + 768;

    // 0) zero bev + stats
    zero_kernel<<<8192, 256, 0, stream>>>((U128*)bev, 2097152);
    zero_kernel<<<1, 256, 0, stream>>>((U128*)stats, 256);
    // 0b) weight pre-transpose to bf16 (o, tap, cin)
    convert_w3_kernel<<<2304, 256, 0, stream>>>(convA_w, wtA);
    convert_w3_kernel<<<2304, 256, 0, stream>>>(convB_w, wtB);
    // 1) PFN (layer2 on WMMA) + pool + scatter
    pfn_wmma_kernel<<<NVOX / 4, 128, 0, stream>>>(
        pillars, num_points, indices, w1, b1, w2, b2, bev);
    // 2) 1x1 conv 64->256
    conv1x1_wmma<<<8192, 128, 0, stream>>>(bev, conv1_w, conv1_b, t1);
    // 3) convA 3x3 + stats
    conv3x3_wmma<<<8192, 128, 0, stream>>>(t1, wtA, convA_b, tA, sumA, sqA);
    // 4) BN-A + ReLU (in place on tA)
    bn_relu_bf16_kernel<<<32768, 256, 0, stream>>>(tA, tA, sumA, sqA, bnA_g, bnA_b);
    // 5) convB 3x3 + stats (into t1)
    conv3x3_wmma<<<8192, 128, 0, stream>>>(tA, wtB, convB_b, t1, sumB, sqB);
    // 6) BN-B + ReLU -> NCHW f32 output
    bn_relu_final_kernel<<<8192, 256, 0, stream>>>(t1, (float*)d_out, sumB, sqB, bnB_g, bnB_b);
}
